// Topographical_Attention_4509715660954
// MI455X (gfx1250) — matrix operand order repaired and checked by reading, not compiled, not run on hardware
//
#include <hip/hip_runtime.h>

#define N_ 64
#define C_ 128
#define F_ 2048
#define PITCH 136  // bf16 elements per LDS row (128 + 8 pad, keeps 16B alignment)

typedef __attribute__((ext_vector_type(16))) __bf16 v16bf;
typedef __attribute__((ext_vector_type(8)))  __bf16 v8bf;
typedef __attribute__((ext_vector_type(4)))  __bf16 v4bf;
typedef __attribute__((ext_vector_type(8)))  float  v8f;

// 16 contiguous fp32 -> bf16 fragment (B operand: K ascending within lane-half)
static __device__ __forceinline__ v16bf cvt_contig16(const float* __restrict__ p) {
  const float4 x0 = *(const float4*)(p + 0);
  const float4 x1 = *(const float4*)(p + 4);
  const float4 x2 = *(const float4*)(p + 8);
  const float4 x3 = *(const float4*)(p + 12);
  v16bf r;
  r[ 0] = (__bf16)x0.x; r[ 1] = (__bf16)x0.y; r[ 2] = (__bf16)x0.z; r[ 3] = (__bf16)x0.w;
  r[ 4] = (__bf16)x1.x; r[ 5] = (__bf16)x1.y; r[ 6] = (__bf16)x1.z; r[ 7] = (__bf16)x1.w;
  r[ 8] = (__bf16)x2.x; r[ 9] = (__bf16)x2.y; r[10] = (__bf16)x2.z; r[11] = (__bf16)x2.w;
  r[12] = (__bf16)x3.x; r[13] = (__bf16)x3.y; r[14] = (__bf16)x3.z; r[15] = (__bf16)x3.w;
  return r;
}

// A operand per ISA layout: elems 0..7 = p[0..7] (K=+0..7), elems 8..15 = p[16..23]
static __device__ __forceinline__ v16bf cvt_split8(const float* __restrict__ p) {
  const float4 x0 = *(const float4*)(p + 0);
  const float4 x1 = *(const float4*)(p + 4);
  const float4 x2 = *(const float4*)(p + 16);
  const float4 x3 = *(const float4*)(p + 20);
  v16bf r;
  r[ 0] = (__bf16)x0.x; r[ 1] = (__bf16)x0.y; r[ 2] = (__bf16)x0.z; r[ 3] = (__bf16)x0.w;
  r[ 4] = (__bf16)x1.x; r[ 5] = (__bf16)x1.y; r[ 6] = (__bf16)x1.z; r[ 7] = (__bf16)x1.w;
  r[ 8] = (__bf16)x2.x; r[ 9] = (__bf16)x2.y; r[10] = (__bf16)x2.z; r[11] = (__bf16)x2.w;
  r[12] = (__bf16)x3.x; r[13] = (__bf16)x3.y; r[14] = (__bf16)x3.z; r[15] = (__bf16)x3.w;
  return r;
}

// --------------------------------------------------------------------------
// Pass 1: scores[n][c][m] = sum_f X[n,c,f] * A[c,m,f]
// One workgroup per c. Wave w owns m-tile w (16 cols), accumulates 4 n-tiles.
// --------------------------------------------------------------------------
__global__ __launch_bounds__(256) void topo_scores_kernel(
    const float* __restrict__ X, const float* __restrict__ A,
    float* __restrict__ S) {
  const int c     = blockIdx.x;
  const int w     = threadIdx.x >> 5;
  const int lane  = threadIdx.x & 31;
  const int lhalf = lane >> 4;   // 0: lanes 0-15, 1: lanes 16-31
  const int l16   = lane & 15;

  const int mrow = w * 16 + l16;                               // B column
  const float* __restrict__ Ab =
      A + ((size_t)c * C_ + mrow) * F_ + lhalf * 8;            // B frag base
  const float* __restrict__ xrow[4];
#pragma unroll
  for (int nt = 0; nt < 4; ++nt)
    xrow[nt] = X + ((size_t)(nt * 16 + l16) * C_ + c) * F_ + lhalf * 8;

  v8f acc[4] = {};
  for (int k0 = 0; k0 < F_; k0 += 32) {
    __builtin_prefetch(Ab + k0 + 512, 0, 0);   // global_prefetch_b8, A stream
    const v16bf b = cvt_split8(Ab + k0);
#pragma unroll
    for (int nt = 0; nt < 4; ++nt) {
      const v16bf a = cvt_split8(xrow[nt] + k0);
      acc[nt] = __builtin_amdgcn_wmma_f32_16x16x32_bf16(
          false, a, false, b, (short)0, acc[nt], false, false);
    }
  }

#pragma unroll
  for (int nt = 0; nt < 4; ++nt)
#pragma unroll
    for (int r = 0; r < 8; ++r) {
      const int n = nt * 16 + r + lhalf * 8;
      S[((size_t)n * C_ + c) * C_ + mrow] = acc[nt][r];
    }
}

// --------------------------------------------------------------------------
// Pass 2: row softmax over m (128), fp32 in -> bf16 out. One wave per row.
// --------------------------------------------------------------------------
__global__ __launch_bounds__(256) void topo_softmax_kernel(
    const float* __restrict__ S, __bf16* __restrict__ W) {
  const int row  = (int)((blockIdx.x * blockDim.x + threadIdx.x) >> 5);
  const int lane = threadIdx.x & 31;
  const float4 x = *(const float4*)(S + (size_t)row * C_ + lane * 4);

  float m = fmaxf(fmaxf(x.x, x.y), fmaxf(x.z, x.w));
#pragma unroll
  for (int off = 16; off > 0; off >>= 1) m = fmaxf(m, __shfl_xor(m, off, 32));

  const float e0 = __expf(x.x - m), e1 = __expf(x.y - m);
  const float e2 = __expf(x.z - m), e3 = __expf(x.w - m);
  float sum = e0 + e1 + e2 + e3;
#pragma unroll
  for (int off = 16; off > 0; off >>= 1) sum += __shfl_xor(sum, off, 32);
  const float inv = 1.0f / sum;

  v4bf o;
  o[0] = (__bf16)(e0 * inv); o[1] = (__bf16)(e1 * inv);
  o[2] = (__bf16)(e2 * inv); o[3] = (__bf16)(e3 * inv);
  *(v4bf*)(W + (size_t)row * C_ + lane * 4) = o;
}

// --------------------------------------------------------------------------
// Pass 3: out[n] = W[n] (128x128) @ X[n] (128x2048), per (n, 128-wide f tile).
// X tile staged transposed (f rows, m cols) in LDS as bf16 for B fragments.
// --------------------------------------------------------------------------
__global__ __launch_bounds__(256) void topo_combine_kernel(
    const float* __restrict__ X, const __bf16* __restrict__ Wt,
    float* __restrict__ O) {
  __shared__ __bf16 lds[128 * PITCH];

  const int n  = blockIdx.y;
  const int f0 = blockIdx.x * 128;
  const float*  __restrict__ Xn = X  + (size_t)n * C_ * F_;
  const __bf16* __restrict__ Wn = Wt + (size_t)n * C_ * C_;
  float*        __restrict__ On = O  + (size_t)n * C_ * F_;
  const int tid = threadIdx.x;

  // Stage: lds[f_local][m] = bf16(X[n][m][f0 + f_local])
  {
    const int fc = (tid & 31) * 4;
    for (int m = tid >> 5; m < C_; m += 8) {
      const float4 x = *(const float4*)(Xn + (size_t)m * F_ + f0 + fc);
      lds[(fc + 0) * PITCH + m] = (__bf16)x.x;
      lds[(fc + 1) * PITCH + m] = (__bf16)x.y;
      lds[(fc + 2) * PITCH + m] = (__bf16)x.z;
      lds[(fc + 3) * PITCH + m] = (__bf16)x.w;
    }
  }
  __syncthreads();

  const int w = tid >> 5, lane = tid & 31;
  const int lhalf = lane >> 4, l16 = lane & 15;
  const int crow = w * 16 + l16;   // output row (c) / A-matrix row

  v8f acc[8] = {};
  for (int k0 = 0; k0 < C_; k0 += 32) {
    // A fragment: W[n][crow][k0 + lhalf*8 + {0..7, 16..23}], bf16 from global
    const __bf16* __restrict__ wr = Wn + (size_t)crow * C_ + k0 + lhalf * 8;
    const v8bf wlo = *(const v8bf*)(wr);
    const v8bf whi = *(const v8bf*)(wr + 16);
    const v16bf a = __builtin_shufflevector(
        wlo, whi, 0, 1, 2, 3, 4, 5, 6, 7, 8, 9, 10, 11, 12, 13, 14, 15);

    const int kb = k0 + lhalf * 8;
#pragma unroll
    for (int t = 0; t < 8; ++t) {
      const __bf16* __restrict__ bp = &lds[(t * 16 + l16) * PITCH + kb];
      const v8bf b0 = *(const v8bf*)(bp);
      const v8bf b1 = *(const v8bf*)(bp + 16);
      const v16bf b = __builtin_shufflevector(
          b0, b1, 0, 1, 2, 3, 4, 5, 6, 7, 8, 9, 10, 11, 12, 13, 14, 15);
      acc[t] = __builtin_amdgcn_wmma_f32_16x16x32_bf16(
          false, a, false, b, (short)0, acc[t], false, false);
    }
  }

#pragma unroll
  for (int t = 0; t < 8; ++t)
#pragma unroll
    for (int r = 0; r < 8; ++r) {
      const int c = w * 16 + r + lhalf * 8;
      const int f = f0 + t * 16 + l16;
      On[(size_t)c * F_ + f] = acc[t][r];
    }
}

extern "C" void kernel_launch(void* const* d_in, const int* in_sizes, int n_in,
                              void* d_out, int out_size, void* d_ws, size_t ws_size,
                              hipStream_t stream) {
  (void)in_sizes; (void)n_in; (void)out_size; (void)ws_size;
  const float* X = (const float*)d_in[0];   // [N, C, F] fp32
  const float* A = (const float*)d_in[1];   // [C, C, F] fp32
  float* out = (float*)d_out;               // [N, C, F] fp32

  float*  S  = (float*)d_ws;                                        // 4 MB fp32 scores
  __bf16* Wb = (__bf16*)((char*)d_ws + (size_t)N_ * C_ * C_ * 4);   // 2 MB bf16 softmax

  topo_scores_kernel <<<dim3(C_),           256, 0, stream>>>(X, A, S);
  topo_softmax_kernel<<<dim3((N_ * C_) / 8), 256, 0, stream>>>(S, Wb);
  topo_combine_kernel<<<dim3(F_ / 128, N_), 256, 0, stream>>>(X, Wb, out);
}
